// EnhancedOFTLinearLayer_46411416601017
// MI455X (gfx1250) — matrix operand order repaired and checked by reading, not compiled, not run on hardware
//
#include <hip/hip_runtime.h>
#include <stdint.h>

#define IN_F  4096
#define OUT_F 4096
#define NB    8
#define BS    512
#define SEQ   (4 * 2048) /* 8192 rows of x */

typedef float  v2f   __attribute__((ext_vector_type(2)));
typedef float  v8f   __attribute__((ext_vector_type(8)));
typedef __bf16 v8bf  __attribute__((ext_vector_type(8)));
typedef __bf16 v16bf __attribute__((ext_vector_type(16)));

#if __has_builtin(__builtin_amdgcn_global_load_async_to_lds_b128) && \
    __has_builtin(__builtin_amdgcn_s_wait_asynccnt)
#define OFT_HAVE_ASYNC_LDS 1
// builtin signature (from hipcc diagnostic): param 1 is a pointer to a
// 4 x i32 vector in the global address space; param 2 the LDS counterpart.
typedef int oft_v4i __attribute__((vector_size(16)));
typedef __attribute__((address_space(1))) oft_v4i* oft_gv4i;
typedef __attribute__((address_space(3))) oft_v4i* oft_lv4i;
#endif

// ---------------------------------------------------------------------------
// 1) M = I + 0.5(A - A^T) + 1e-6 I ;  N = I - 0.5(A - A^T)
// ---------------------------------------------------------------------------
__global__ void oft_skew_prep(const float* __restrict__ R,
                              float* __restrict__ Mmat,
                              float* __restrict__ Nmat)
{
    int idx = blockIdx.x * blockDim.x + threadIdx.x;
    if (idx >= NB * BS * BS) return;
    int n   = idx / (BS * BS);
    int rem = idx - n * BS * BS;
    int r   = rem / BS;
    int c   = rem - r * BS;
    const float* A = R + (size_t)n * BS * BS;
    float s   = 0.5f * (A[r * BS + c] - A[c * BS + r]);
    float eye = (r == c) ? 1.0f : 0.0f;
    Mmat[idx] = eye * (1.0f + 1e-6f) + s;
    Nmat[idx] = eye - s;
}

// ---------------------------------------------------------------------------
// 2) In-place Gauss-Jordan inverse, one workgroup per 512x512 matrix.
//    I + skew is always invertible & well conditioned -> no pivoting.
//    Pivot row is staged to LDS with ASYNCcnt-tracked async copies when the
//    toolchain exposes GLOBAL_LOAD_ASYNC_TO_LDS.
// ---------------------------------------------------------------------------
__global__ __launch_bounds__(1024) void oft_gauss_jordan(float* __restrict__ Mall)
{
    float* A = Mall + (size_t)blockIdx.x * BS * BS;
    __shared__ __align__(16) float praw[BS];  // raw pivot row (async target)
    __shared__ float prow[BS];                // normalized pivot row
    __shared__ float pinv_s;
    const int tid  = threadIdx.x;
    const int wid  = tid >> 5;   // 32 waves
    const int lane = tid & 31;

    for (int k = 0; k < BS; ++k) {
        // ---- stage raw pivot row k into LDS ----
#if defined(OFT_HAVE_ASYNC_LDS)
        if (tid < BS / 4) {  // 128 lanes x b128 = 2 KB row
            float* g = A + (size_t)k * BS + tid * 4;
            __builtin_amdgcn_global_load_async_to_lds_b128(
                (oft_gv4i)(void*)g, (oft_lv4i)(void*)&praw[tid * 4], 0, 0);
            __builtin_amdgcn_s_wait_asynccnt(0);
        }
#else
        for (int c = tid; c < BS; c += 1024) praw[c] = A[(size_t)k * BS + c];
#endif
        __syncthreads();
        if (tid == 0) pinv_s = 1.0f / praw[k];
        __syncthreads();
        const float pinv = pinv_s;

        // ---- normalize pivot row (LDS + writeback) ----
        for (int c = tid; c < BS; c += 1024) {
            float v = (c == k) ? pinv : praw[c] * pinv;
            A[(size_t)k * BS + c] = v;
            prow[c] = v;
        }
        __syncthreads();

        // ---- warp-per-row elimination: scalar f is read by the whole wave
        //      before any lane of that wave writes column k (wave32 lockstep)
        for (int r = wid; r < BS; r += 32) {
            if (r == k) continue;
            const float f = A[(size_t)r * BS + k];
            for (int c = lane; c < BS; c += 32) {
                float v = (c == k) ? (-f * pinv) : (A[(size_t)r * BS + c] - f * prow[c]);
                A[(size_t)r * BS + c] = v;
            }
        }
        __syncthreads();
    }
}

// ---------------------------------------------------------------------------
// 3/4) Batched f32 WMMA GEMM, C = A @ B (NN), one 16x16 tile per wave.
//      Native v_wmma_f32_16x16x4_f32 keeps the Cayley/rotation stages at full
//      f32 precision (they feed everything downstream, only ~21 GFLOP).
//      Pointer-increment loop so no per-iteration 64-bit address multiplies.
// ---------------------------------------------------------------------------
#define GEMM_WAVES 4
__global__ __launch_bounds__(32 * GEMM_WAVES) void oft_wmma_gemm_f32_nn(
    const float* __restrict__ A, const float* __restrict__ B, float* __restrict__ C,
    int K, int lda, int ldb, int ldc,
    long long aBatch, long long bBatch, long long cBatch)
{
    const int wave   = threadIdx.x >> 5;
    const int lane   = threadIdx.x & 31;
    const int tile_n = blockIdx.x * GEMM_WAVES + wave;
    const int tile_m = blockIdx.y;
    const int batch  = blockIdx.z;
    A += (size_t)batch * aBatch;
    B += (size_t)batch * bBatch;
    C += (size_t)batch * cBatch;

    // f32 A 16x4 layout: lanes 0-15 hold K=0,1 ; lanes 16-31 hold K=2,3
    const int row = tile_m * 16 + (lane & 15);
    const int col = tile_n * 16 + (lane & 15);
    const int kh  = (lane >> 4) * 2;

    const float* ap = A + (size_t)row * lda + kh;
    const float* bp = B + (size_t)kh * ldb + col;
    const size_t bstep = (size_t)4 * ldb;

    v8f acc = {};
    for (int k0 = 0; k0 < K; k0 += 4) {
        v2f a, b;
        a.x = ap[0];
        a.y = ap[1];
        b.x = bp[0];
        b.y = bp[ldb];
        acc = __builtin_amdgcn_wmma_f32_16x16x4_f32(false, a, false, b,
                                                    (short)0, acc, false, false);
        ap += 4;
        bp += bstep;
    }
    // C/D layout: VGPR r -> M = r + (lane>=16 ? 8 : 0), N = lane%16
    const int srow = tile_m * 16 + (lane >> 4) * 8;
#pragma unroll
    for (int r = 0; r < 8; ++r)
        C[(size_t)(srow + r) * ldc + col] = acc[r];
}

// ---------------------------------------------------------------------------
// 5) out = x @ tw^T + bias  (8192 x 4096 x 4096), bf16x3 split precision.
//    64x64 block tile, K-panel 32 staged in LDS as hi/lo bf16 (split done on
//    the fly -> no extra HBM traffic). A panel stored in natural k
//    order (same as B) so BOTH A and B fragments are single
//    contiguous 32-byte LDS vector loads; staging uses b128 LDS stores.
//    8 waves x 2 n-tiles, 6 bf16 WMMAs per wave per panel = f32-class
//    accuracy at 3/8 the instruction count of native f32 WMMA.
// ---------------------------------------------------------------------------
__global__ __launch_bounds__(256) void oft_gemm_bf16x3(
    const float* __restrict__ X,    // [8192, 4096] row-major
    const float* __restrict__ TW,   // [4096, 4096] row-major (rows = out feature)
    const float* __restrict__ bias, // [4096]
    float* __restrict__ Out)        // [8192, 4096]
{
    constexpr int K = IN_F;
    constexpr int N = OUT_F;
    // A panels stored in natural k order (same as B)
    __shared__ __align__(32) __bf16 as_hi[64][32];
    __shared__ __align__(32) __bf16 as_lo[64][32];
    __shared__ __align__(32) __bf16 bs_hi[64][32];  // natural k order
    __shared__ __align__(32) __bf16 bs_lo[64][32];

    const int tid   = threadIdx.x;
    const int lane  = tid & 31;
    const int wave  = tid >> 5;
    const int mBase = blockIdx.y * 64;
    const int nBase = blockIdx.x * 64;

    // staging role: thread -> (row 0..63, 8 consecutive k)
    const int srow = tid >> 2;
    const int sk   = (tid & 3) * 8;                              // k base: 0,8,16,24
    const int spA  = (sk & 7) | ((sk & 8) << 1) | ((sk & 16) >> 1); // permuted: 0,16,8,24
    const float* xrow = X  + (size_t)(mBase + srow) * K + sk;
    const float* trow = TW + (size_t)(nBase + srow) * K + sk;

    // compute role: wave -> (m-tile, pair of n-tiles)
    const int wm    = wave & 3;       // 4 m-tiles of 16
    const int wn    = wave >> 2;      // 2 n-groups of 32
    const int tm    = wm * 16;
    const int n0    = wn * 32;
    const int arow  = tm + (lane & 15);
    const int brow0 = n0 + (lane & 15);
    const int fo    = (lane >> 4) * 16;  // fragment element offset (A & B)

    v8f acc0 = {};
    v8f acc1 = {};

    for (int k0 = 0; k0 < K; k0 += 32) {
        // load next panel to registers before the barrier (pipelining-lite)
        float4 xa = *(const float4*)(xrow + k0);
        float4 xb = *(const float4*)(xrow + k0 + 4);
        float4 ta = *(const float4*)(trow + k0);
        float4 tb = *(const float4*)(trow + k0 + 4);
        if (k0 + 32 < K) {
            __builtin_prefetch(xrow + k0 + 32, 0, 3);  // global_prefetch_b8
            __builtin_prefetch(trow + k0 + 32, 0, 3);
        }
        __syncthreads();  // previous panel fully consumed

        float xv[8] = {xa.x, xa.y, xa.z, xa.w, xb.x, xb.y, xb.z, xb.w};
        float tv[8] = {ta.x, ta.y, ta.z, ta.w, tb.x, tb.y, tb.z, tb.w};
        v8bf ahv, alv, bhv, blv;
#pragma unroll
        for (int j = 0; j < 8; ++j) {
            __bf16 h = (__bf16)xv[j];
            ahv[j] = h;
            alv[j] = (__bf16)(xv[j] - (float)h);
            __bf16 g = (__bf16)tv[j];
            bhv[j] = g;
            blv[j] = (__bf16)(tv[j] - (float)g);
        }
        // one 16-byte LDS store per array per thread (ds_store_b128)
        *(v8bf*)&as_hi[srow][sk] = ahv;
        *(v8bf*)&as_lo[srow][sk] = alv;
        *(v8bf*)&bs_hi[srow][sk]  = bhv;
        *(v8bf*)&bs_lo[srow][sk]  = blv;
        __syncthreads();

        // fragments: contiguous 32-byte LDS vector loads per lane
        // A (16x32): lanes<16 hold K=0..15 of row lane, lanes>=16 K=16..31
        // (contiguous k order, matching B).
        v16bf ah = *(const v16bf*)&as_hi[arow][fo];
        v16bf al = *(const v16bf*)&as_lo[arow][fo];
        // B (32x16): lanes<16 hold K=0..15 of column lane, lanes>=16 K=16..31
        v16bf bh0 = *(const v16bf*)&bs_hi[brow0][fo];
        v16bf bl0 = *(const v16bf*)&bs_lo[brow0][fo];
        v16bf bh1 = *(const v16bf*)&bs_hi[brow0 + 16][fo];
        v16bf bl1 = *(const v16bf*)&bs_lo[brow0 + 16][fo];

        // hi*hi + hi*lo + lo*hi, interleaved across the two accumulators
        acc0 = __builtin_amdgcn_wmma_f32_16x16x32_bf16(false, ah, false, bh0, (short)0, acc0, false, false);
        acc1 = __builtin_amdgcn_wmma_f32_16x16x32_bf16(false, ah, false, bh1, (short)0, acc1, false, false);
        acc0 = __builtin_amdgcn_wmma_f32_16x16x32_bf16(false, ah, false, bl0, (short)0, acc0, false, false);
        acc1 = __builtin_amdgcn_wmma_f32_16x16x32_bf16(false, ah, false, bl1, (short)0, acc1, false, false);
        acc0 = __builtin_amdgcn_wmma_f32_16x16x32_bf16(false, al, false, bh0, (short)0, acc0, false, false);
        acc1 = __builtin_amdgcn_wmma_f32_16x16x32_bf16(false, al, false, bh1, (short)0, acc1, false, false);
    }

    // epilogue: C layout VGPR r -> M = r + (lane>=16 ? 8 : 0), N = lane%16
    const int gm = mBase + tm + (lane >> 4) * 8;
    {
        const int gn = nBase + n0 + (lane & 15);
        const float bv = bias[gn];
#pragma unroll
        for (int r = 0; r < 8; ++r)
            Out[(size_t)(gm + r) * N + gn] = acc0[r] + bv;
    }
    {
        const int gn = nBase + n0 + 16 + (lane & 15);
        const float bv = bias[gn];
#pragma unroll
        for (int r = 0; r < 8; ++r)
            Out[(size_t)(gm + r) * N + gn] = acc1[r] + bv;
    }
}

// ---------------------------------------------------------------------------
extern "C" void kernel_launch(void* const* d_in, const int* in_sizes, int n_in,
                              void* d_out, int out_size, void* d_ws, size_t ws_size,
                              hipStream_t stream)
{
    (void)in_sizes; (void)n_in; (void)out_size; (void)ws_size;
    const float* weight = (const float*)d_in[0];  // [4096, 4096]
    const float* bias   = (const float*)d_in[1];  // [4096]
    const float* x      = (const float*)d_in[2];  // [4, 2048, 4096]
    const float* Rb     = (const float*)d_in[3];  // [8, 512, 512]
    // d_in[4] = layer_idx (unused by the math)
    float* out = (float*)d_out;                   // [4, 2048, 4096]

    // workspace layout (floats): M, N, Q (8*512*512 each) + tw (4096*4096)
    float* Mws = (float*)d_ws;
    float* Nws = Mws + (size_t)NB * BS * BS;
    float* Qws = Nws + (size_t)NB * BS * BS;
    float* TW  = Qws + (size_t)NB * BS * BS;

    // 1) skew prep
    {
        int total = NB * BS * BS;
        oft_skew_prep<<<(total + 255) / 256, 256, 0, stream>>>(Rb, Mws, Nws);
    }
    // 2) invert M in place (one WG per block matrix)
    oft_gauss_jordan<<<NB, 1024, 0, stream>>>(Mws);
    // 3) Q = (I - S) @ (I + S)^-1   — batched over z
    {
        dim3 grid(BS / 16 / GEMM_WAVES, BS / 16, NB);
        oft_wmma_gemm_f32_nn<<<grid, 32 * GEMM_WAVES, 0, stream>>>(
            Nws, Mws, Qws, BS, BS, BS, BS,
            (long long)BS * BS, (long long)BS * BS, (long long)BS * BS);
    }
    // 4) tw[:, n*BS:(n+1)*BS] = weight[:, n*BS:(n+1)*BS] @ Q_n
    {
        dim3 grid(BS / 16 / GEMM_WAVES, OUT_F / 16, NB);
        oft_wmma_gemm_f32_nn<<<grid, 32 * GEMM_WAVES, 0, stream>>>(
            weight, Qws, TW, BS, IN_F, BS, IN_F,
            (long long)BS, (long long)BS * BS, (long long)BS);
    }
    // 5) out = x @ tw^T + bias   (bf16x3 split, dominant 275 GFLOP)
    {
        dim3 grid(OUT_F / 64, SEQ / 64, 1);
        oft_gemm_bf16x3<<<grid, 256, 0, stream>>>(x, TW, bias, out);
    }
}
